// GraphAttentionLayer_43602507989722
// MI455X (gfx1250) — matrix operand order repaired and checked by reading, not compiled, not run on hardware
//
#include <hip/hip_runtime.h>
#include <hip/hip_bf16.h>

typedef _Float16 v16h __attribute__((ext_vector_type(16)));
typedef _Float16 h8   __attribute__((ext_vector_type(8)));
typedef float    v8f  __attribute__((ext_vector_type(8)));
typedef float    f8v  __attribute__((ext_vector_type(8)));
typedef unsigned int v4u __attribute__((ext_vector_type(4)));
typedef int          v8i __attribute__((ext_vector_type(8)));
typedef int          v4i __attribute__((ext_vector_type(4)));

#define NHEAD 8
#define NN    4096
#define FF    128

#if defined(__has_builtin)
#if __has_builtin(__builtin_amdgcn_tensor_load_to_lds)
#define USE_TDM 1
#endif
#endif
#ifndef USE_TDM
#define USE_TDM 0
#endif

// ---------------------------------------------------------------- converts
__global__ __launch_bounds__(256) void k_cvt_h(const float* __restrict__ h,
                                               _Float16* __restrict__ h16) {
  int idx = blockIdx.x * 256 + threadIdx.x;          // 4096*128
  h16[idx] = (_Float16)h[idx];
}

__global__ __launch_bounds__(256) void k_build_wT(const float* __restrict__ w,
                                                  _Float16* __restrict__ wT) {
  int idx  = blockIdx.x * 256 + threadIdx.x;         // 8*128*128
  int head = idx >> 14;
  int rem  = idx & 16383;
  int fi   = rem >> 7;                               // input feature (K)
  int fo   = rem & 127;                              // output feature (N)
  wT[head * 16384 + fo * 128 + fi] = (_Float16)w[idx];
}

__global__ void k_init_dmax(unsigned int* dmax_enc) {
  if (threadIdx.x < NHEAD) dmax_enc[threadIdx.x] = 0x007FFFFFu;  // enc(-inf)
}

// ------------------------------------------------- stage 1: h' = h @ w[h]
__global__ __launch_bounds__(256) void k_hprime(const _Float16* __restrict__ h16,
                                                const _Float16* __restrict__ wT16,
                                                _Float16* __restrict__ hp_t) {
  const int head = blockIdx.y;
  const int i0   = blockIdx.x * 128;
  const int wv   = threadIdx.x >> 5;
  const int lane = threadIdx.x & 31;
  const int lm   = lane & 15;
  const int lh   = lane >> 4;
  const int row  = i0 + wv * 16 + lm;
  const _Float16* wTh = wT16 + head * (FF * FF);

  v8f acc[8] = {};
  #pragma unroll
  for (int ks = 0; ks < 4; ++ks) {
    const int kb = ks * 32;
    h8 alo = *(const h8*)&h16[row * FF + kb + lh * 8];
    h8 ahi = *(const h8*)&h16[row * FF + kb + lh * 8 + 16];
    v16h A;
    #pragma unroll
    for (int e = 0; e < 8; ++e) { A[e] = alo[e]; A[e + 8] = ahi[e]; }
    #pragma unroll
    for (int t = 0; t < 8; ++t) {
      const int n = t * 16 + lm;
      const _Float16* bp = &wTh[n * FF + kb + lh * 8];
      h8 blo = *(const h8*)bp;
      h8 bhi = *(const h8*)(bp + 16);
      v16h B;
      #pragma unroll
      for (int e = 0; e < 8; ++e) { B[e] = blo[e]; B[e + 8] = bhi[e]; }
      acc[t] = __builtin_amdgcn_wmma_f32_16x16x32_f16(false, A, false, B,
                                                      (short)0, acc[t], false, false);
    }
  }
  const int j0 = i0 + wv * 16 + lh * 8;
  #pragma unroll
  for (int t = 0; t < 8; ++t) {
    const int f = t * 16 + lm;
    h8 outv;
    #pragma unroll
    for (int r = 0; r < 8; ++r) outv[r] = (_Float16)acc[t][r];
    *(h8*)&hp_t[(size_t)head * FF * NN + (size_t)f * NN + j0] = outv;
  }
}

// ---------------------- stage 2: s,d,E1,E2 and per-head D = max_j d_j
__global__ __launch_bounds__(256) void k_prep(const _Float16* __restrict__ hp_t,
                                              const float* __restrict__ a_src,
                                              const float* __restrict__ a_dst,
                                              float* __restrict__ s_out,
                                              float* __restrict__ e1,
                                              float* __restrict__ e2,
                                              unsigned int* __restrict__ dmax_enc) {
  const int head = blockIdx.y;
  const int i    = blockIdx.x * 256 + threadIdx.x;
  const _Float16* hp = hp_t + (size_t)head * FF * NN;
  float s = 0.f, d = 0.f;
  #pragma unroll 8
  for (int f = 0; f < FF; ++f) {
    float v = (float)hp[(size_t)f * NN + i];
    s += v * a_src[head * FF + f];
    d += v * a_dst[head * FF + f];
  }
  s_out[head * NN + i] = s;
  e1[head * NN + i] = __expf(d);
  e2[head * NN + i] = __expf(0.2f * d);

  __shared__ float red[256];
  red[threadIdx.x] = d;
  __syncthreads();
  for (int off = 128; off > 0; off >>= 1) {
    if (threadIdx.x < off)
      red[threadIdx.x] = fmaxf(red[threadIdx.x], red[threadIdx.x + off]);
    __syncthreads();
  }
  if (threadIdx.x == 0) {
    unsigned int b = __float_as_uint(red[0]);
    unsigned int enc = (b & 0x80000000u) ? ~b : (b | 0x80000000u);
    atomicMax(&dmax_enc[head], enc);
  }
}

#if USE_TDM
// Issue TDM load of one 32(j) x 128(f) f16 tile of h'^T into LDS.
// D# per CDNA5 ISA: addr = ga + 2*(x + y*4096)
__device__ __forceinline__ void tdm_issue_tile(const _Float16* gbase, unsigned lds_off) {
  unsigned long long ga = (unsigned long long)(size_t)gbase;
  v4u g0;
  g0[0] = 1u;                                            // count=1 (valid user D#)
  g0[1] = lds_off;                                       // lds_addr (bytes)
  g0[2] = (unsigned)(ga & 0xFFFFFFFFu);                  // global_addr[31:0]
  g0[3] = (unsigned)((ga >> 32) & 0x01FFFFFFu)           // global_addr[56:32]
          | 0x80000000u;                                 // type=2 ("image")
  v8i g1;
  g1[0] = 0x00010000;                                    // wg_mask=0, data_size=2B
  g1[1] = (int)(4096u << 16);                            // tensor_dim0 = 4096
  g1[2] = (int)(128u  << 16);                            // tensor_dim1 = 128
  g1[3] = (int)(32u   << 16);                            // tile_dim0   = 32
  g1[4] = 128;                                           // tile_dim1   = 128
  g1[5] = 4096;                                          // tensor_dim0_stride = 4096
  g1[6] = 0;
  g1[7] = 0;
  v4i gz4 = {0, 0, 0, 0};                                // groups 2/3: 2D tile
  v8i gz8 = {0, 0, 0, 0, 0, 0, 0, 0};                    // extra group (clang-23 form)
  __builtin_amdgcn_tensor_load_to_lds(g0, g1, gz4, gz4, gz8, 0);
}
#endif

// ------------- stage 3: out_h = softmax(LR(s_i+d_j)) @ h'  (flash-style)
__global__ __launch_bounds__(256) void k_attn(const _Float16* __restrict__ hp_t,
                                              const float* __restrict__ s_arr,
                                              const float* __restrict__ e1,
                                              const float* __restrict__ e2,
                                              const unsigned int* __restrict__ dmax_enc,
                                              float* __restrict__ po) {
  const int head = blockIdx.y;
  const int i0   = blockIdx.x * 128;
  const int wv   = threadIdx.x >> 5;
  const int lane = threadIdx.x & 31;
  const int lm   = lane & 15;
  const int lh   = lane >> 4;
  const int row  = i0 + wv * 16 + lm;

  unsigned int de = dmax_enc[head];
  float D = __uint_as_float((de & 0x80000000u) ? (de & 0x7fffffffu) : ~de);
  float s  = s_arr[head * NN + row];
  float t0 = s + D;
  float m  = (t0 > 0.f) ? t0 : 0.2f * t0;
  float P  = __expf(s - m);
  float Q  = __expf(0.2f * s - m);
  float R  = __expf(-s);

  const float*    E1 = e1 + head * NN;
  const float*    E2 = e2 + head * NN;
  const _Float16* hp = hp_t + (size_t)head * FF * NN;

  __shared__ _Float16 lds[2][FF * 32];                   // double-buffered 8KB tiles

  v8f acc[8] = {};
  float zsum = 0.f;

#if USE_TDM
  const unsigned lds_off0 = (unsigned)(size_t)&lds[0][0];  // LDS aperture: offset in [31:0]
  const unsigned lds_off1 = (unsigned)(size_t)&lds[1][0];
  if (wv == 0) tdm_issue_tile(hp, lds_off0);             // prefetch tile 0
#else
  const int tf = threadIdx.x >> 1;
  const int tj = (threadIdx.x & 1) * 16;
#endif

  for (int jb = 0; jb < NN; jb += 32) {
    const int buf = (jb >> 5) & 1;
#if USE_TDM
    if (wv == 0) {
      if (jb + 32 < NN) {                                // prefetch next into other buf
        tdm_issue_tile(hp + jb + 32, buf ? lds_off0 : lds_off1);
        __builtin_amdgcn_s_wait_tensorcnt((short)1);     // current tile resident
      } else {
        __builtin_amdgcn_s_wait_tensorcnt((short)0);
      }
    }
    __syncthreads();                                     // publish tile to all waves
#else
    __syncthreads();
    *(h8*)&lds[buf][tf * 32 + tj]     = *(const h8*)&hp[(size_t)tf * NN + jb + tj];
    *(h8*)&lds[buf][tf * 32 + tj + 8] = *(const h8*)&hp[(size_t)tf * NN + jb + tj + 8];
    __syncthreads();
#endif

    // Generate softmax-weight A fragment directly in WMMA layout (no exp!)
    f8v e1lo = *(const f8v*)&E1[jb + lh * 8];
    f8v e2lo = *(const f8v*)&E2[jb + lh * 8];
    f8v e1hi = *(const f8v*)&E1[jb + 16 + lh * 8];
    f8v e2hi = *(const f8v*)&E2[jb + 16 + lh * 8];
    v16h A;
    #pragma unroll
    for (int e = 0; e < 8; ++e) {
      float wlo = (e1lo[e] > R) ? P * e1lo[e] : Q * e2lo[e];
      float whi = (e1hi[e] > R) ? P * e1hi[e] : Q * e2hi[e];
      zsum += wlo + whi;
      A[e]     = (_Float16)wlo;
      A[e + 8] = (_Float16)whi;
    }
    #pragma unroll
    for (int t = 0; t < 8; ++t) {
      const int f = t * 16 + lm;
      const _Float16* bp = &lds[buf][f * 32 + lh * 8];
      h8 blo = *(const h8*)bp;
      h8 bhi = *(const h8*)(bp + 16);
      v16h B;
      #pragma unroll
      for (int e = 0; e < 8; ++e) { B[e] = blo[e]; B[e + 8] = bhi[e]; }
      acc[t] = __builtin_amdgcn_wmma_f32_16x16x32_f16(false, A, false, B,
                                                      (short)0, acc[t], false, false);
    }
    __syncthreads();                                     // buf reusable next round
  }

  float z = zsum + __shfl_xor(zsum, 16, 32);
  float rz[8];
  #pragma unroll
  for (int r = 0; r < 8; ++r)
    rz[r] = 1.0f / __shfl(z, lh * 8 + r, 32);

  float* poh = po + (size_t)head * NN * FF;
  #pragma unroll
  for (int t = 0; t < 8; ++t) {
    const int f = t * 16 + lm;
    #pragma unroll
    for (int r = 0; r < 8; ++r) {
      const int orow = i0 + wv * 16 + lh * 8 + r;
      poh[(size_t)orow * FF + f] = acc[t][r] * rz[r];
    }
  }
}

// ----------------------------- stage 4: mean over heads + bias
__global__ __launch_bounds__(256) void k_reduce(const float* __restrict__ po,
                                                const float* __restrict__ bias,
                                                float* __restrict__ out) {
  int idx = blockIdx.x * 256 + threadIdx.x;
  float acc = 0.f;
  #pragma unroll
  for (int hh = 0; hh < NHEAD; ++hh) acc += po[(size_t)hh * NN * FF + idx];
  out[idx] = acc * 0.125f + bias[idx & (FF - 1)];
}

// ---------------------------------------------------------------- launcher
extern "C" void kernel_launch(void* const* d_in, const int* in_sizes, int n_in,
                              void* d_out, int out_size, void* d_ws, size_t ws_size,
                              hipStream_t stream) {
  const float* h     = (const float*)d_in[0];
  const float* w     = (const float*)d_in[1];
  const float* a_src = (const float*)d_in[2];
  const float* a_dst = (const float*)d_in[3];
  const float* bias  = (const float*)d_in[4];
  float* out = (float*)d_out;

  char* ws = (char*)d_ws;
  size_t off = 0;
  _Float16* h16  = (_Float16*)(ws + off); off += (size_t)NN * FF * 2;
  _Float16* wT16 = (_Float16*)(ws + off); off += (size_t)NHEAD * FF * FF * 2;
  _Float16* hp_t = (_Float16*)(ws + off); off += (size_t)NHEAD * FF * NN * 2;
  float* s_buf   = (float*)(ws + off);    off += (size_t)NHEAD * NN * 4;
  float* e1_buf  = (float*)(ws + off);    off += (size_t)NHEAD * NN * 4;
  float* e2_buf  = (float*)(ws + off);    off += (size_t)NHEAD * NN * 4;
  unsigned int* dmax = (unsigned int*)(ws + off); off += 256;
  float* po_buf  = (float*)(ws + off);    off += (size_t)NHEAD * NN * FF * 4;

  k_cvt_h   <<<(NN * FF) / 256, 256, 0, stream>>>(h, h16);
  k_build_wT<<<(NHEAD * FF * FF) / 256, 256, 0, stream>>>(w, wT16);
  k_init_dmax<<<1, 32, 0, stream>>>(dmax);

  k_hprime<<<dim3(NN / 128, NHEAD), 256, 0, stream>>>(h16, wT16, hp_t);
  k_prep  <<<dim3(NN / 256, NHEAD), 256, 0, stream>>>(hp_t, a_src, a_dst,
                                                      s_buf, e1_buf, e2_buf, dmax);
  k_attn  <<<dim3(NN / 128, NHEAD), 256, 0, stream>>>(hp_t, s_buf, e1_buf, e2_buf,
                                                      dmax, po_buf);
  k_reduce<<<(NN * FF) / 256, 256, 0, stream>>>(po_buf, bias, out);
}
